// RGCN_EMB_89240830477002
// MI455X (gfx1250) — matrix operand order repaired and checked by reading, not compiled, not run on hardware
//
#include <hip/hip_runtime.h>

#define NN 50000
#define RR 33
#define RPAD 40      // padded relation count: 8 waves * 5 tiles
#define EE 3200000
#define DD 1600
#define WSZ 16
#define CC 10

typedef __attribute__((ext_vector_type(16))) __bf16 v16bf;
typedef __attribute__((ext_vector_type(8)))  __bf16 v8bf;
typedef __attribute__((ext_vector_type(8)))  float  v8f;

// ---------------------------------------------------------------------------
// Kernel 0: W1[r][i][o] (f32) -> W1t[r][o][i] (bf16); pad rels 33..39 = 0
// ---------------------------------------------------------------------------
__global__ void cvt_w1_kernel(const float* __restrict__ W1, __bf16* __restrict__ W1t) {
    int idx = blockIdx.x * blockDim.x + threadIdx.x;
    if (idx >= RPAD * DD * WSZ) return;
    int r   = idx / (DD * WSZ);
    int rem = idx % (DD * WSZ);
    int i   = rem / WSZ;
    int o   = rem % WSZ;
    float v = (r < RR) ? W1[idx] : 0.0f;
    W1t[((size_t)r * WSZ + o) * DD + i] = (__bf16)v;
}

// ---------------------------------------------------------------------------
// Kernel 1: zero h1
// ---------------------------------------------------------------------------
__global__ void zero_h1_kernel(float* __restrict__ h1) {
    int idx = blockIdx.x * blockDim.x + threadIdx.x;
    if (idx < NN * WSZ) h1[idx] = 0.0f;
}

// ---------------------------------------------------------------------------
// Kernel 2: out[n][c] = bias2[c]
// ---------------------------------------------------------------------------
__global__ void init_out_kernel(float* __restrict__ out, const float* __restrict__ bias2) {
    int idx = blockIdx.x * blockDim.x + threadIdx.x;
    if (idx < NN * CC) out[idx] = bias2[idx % CC];
}

// ---------------------------------------------------------------------------
// helpers for the WMMA kernel
// ---------------------------------------------------------------------------
__device__ __forceinline__ v16bf cvt16(float4 a, float4 b, float4 c, float4 d) {
    v16bf r;
    r[0]  = (__bf16)a.x; r[1]  = (__bf16)a.y; r[2]  = (__bf16)a.z; r[3]  = (__bf16)a.w;
    r[4]  = (__bf16)b.x; r[5]  = (__bf16)b.y; r[6]  = (__bf16)b.z; r[7]  = (__bf16)b.w;
    r[8]  = (__bf16)c.x; r[9]  = (__bf16)c.y; r[10] = (__bf16)c.z; r[11] = (__bf16)c.w;
    r[12] = (__bf16)d.x; r[13] = (__bf16)d.y; r[14] = (__bf16)d.z; r[15] = (__bf16)d.w;
    return r;
}
__device__ __forceinline__ v16bf cat16(v8bf lo, v8bf hi) {
    return __builtin_shufflevector(lo, hi, 0, 1, 2, 3, 4, 5, 6, 7,
                                   8, 9, 10, 11, 12, 13, 14, 15);
}

// ---------------------------------------------------------------------------
// Kernel 3: h[r][n][o] = sum_i X[n][i] * W1[r][i][o]  via bf16 WMMA, f32 accum.
//   grid.x = NN/16 node tiles; block = 256 (8 waves); wave w -> rels w+8t, t<5
//   (relations padded to 40 so the k-loop is branchless).
//   A = W1t tile (M=o, K=i), B = X tile (N=n, K=i), D = h tile.
//   2-stage software pipeline: loads of chunk k+32 overlap WMMAs of chunk k.
// ---------------------------------------------------------------------------
#define RELSTRIDE 204800  /* 8 relations * 16 * DD elements between tiles      */

__global__ __launch_bounds__(256) void gemm1_bf16_kernel(
    const float* __restrict__ X, const __bf16* __restrict__ W1t,
    float* __restrict__ h) {
    const int m0   = blockIdx.x * 16;
    const int lane = threadIdx.x & 31;
    const int wave = threadIdx.x >> 5;
    const int half = lane >> 4;   // 0: lanes 0-15, 1: lanes 16-31
    const int l16  = lane & 15;

    // B operand: lane supplies X row m0+l16; lanes 0-15 K=k..k+15, 16-31 K=k+16..31
    const float*  xrow  = X + (size_t)(m0 + l16) * DD + half * 16;
    // A operand: lane supplies W1t row (o = l16); lanes 0-15 K=k..k+15,
    //            lanes 16-31 K=k+16..31. One base, immediate rel offsets.
    const __bf16* abase = W1t + ((size_t)(wave * 16) + l16) * DD + half * 16;

    v8f acc[5] = {};

    float4 xa0, xa1, xa2, xa3;  v8bf aa[5][2];   // buffer A
    float4 xb0, xb1, xb2, xb3;  v8bf ab[5][2];   // buffer B

    // prologue: chunk k = 0 -> buffer A
    {
        const float4* xp = (const float4*)xrow;
        xa0 = xp[0]; xa1 = xp[1]; xa2 = xp[2]; xa3 = xp[3];
#pragma unroll
        for (int t = 0; t < 5; ++t) {
            aa[t][0] = *(const v8bf*)(abase + t * RELSTRIDE);
            aa[t][1] = *(const v8bf*)(abase + t * RELSTRIDE + 8);
        }
    }

    for (int k = 0; k < DD; k += 64) {
        const int k1 = k + 32;                     // always < DD (DD % 64 == 0 -> k1 max 1568)
        __builtin_prefetch(xrow + k + 256, 0, 0);  // speculative, non-faulting

        // ---- load chunk k1 -> buffer B (overlaps compute below)
        {
            const float4* xp = (const float4*)(xrow + k1);
            xb0 = xp[0]; xb1 = xp[1]; xb2 = xp[2]; xb3 = xp[3];
#pragma unroll
            for (int t = 0; t < 5; ++t) {
                ab[t][0] = *(const v8bf*)(abase + t * RELSTRIDE + k1);
                ab[t][1] = *(const v8bf*)(abase + t * RELSTRIDE + k1 + 8);
            }
        }
        // ---- compute chunk k from buffer A
        {
            v16bf b = cvt16(xa0, xa1, xa2, xa3);
#pragma unroll
            for (int t = 0; t < 5; ++t) {
                acc[t] = __builtin_amdgcn_wmma_f32_16x16x32_bf16(
                    false, cat16(aa[t][0], aa[t][1]), false, b, (short)0,
                    acc[t], false, false);
            }
        }
        // ---- load chunk k+64 (clamped in-bounds) -> buffer A
        int k2 = k + 64;
        if (k2 >= DD) k2 = k;                      // uniform select; redundant safe reload on last trip
        {
            const float4* xp = (const float4*)(xrow + k2);
            xa0 = xp[0]; xa1 = xp[1]; xa2 = xp[2]; xa3 = xp[3];
#pragma unroll
            for (int t = 0; t < 5; ++t) {
                aa[t][0] = *(const v8bf*)(abase + t * RELSTRIDE + k2);
                aa[t][1] = *(const v8bf*)(abase + t * RELSTRIDE + k2 + 8);
            }
        }
        // ---- compute chunk k1 from buffer B
        {
            v16bf b = cvt16(xb0, xb1, xb2, xb3);
#pragma unroll
            for (int t = 0; t < 5; ++t) {
                acc[t] = __builtin_amdgcn_wmma_f32_16x16x32_bf16(
                    false, cat16(ab[t][0], ab[t][1]), false, b, (short)0,
                    acc[t], false, false);
            }
        }
    }

    // D layout: lane<16 -> (n = m0+l16, o = 0..7); lane>=16 -> (n = m0+l16, o = 8..15)
#pragma unroll
    for (int t = 0; t < 5; ++t) {
        int ru = __builtin_amdgcn_readfirstlane(wave + (t << 3));  // uniform branch
        if (ru < RR) {
            float* hp = h + ((size_t)ru * NN + m0 + l16) * WSZ + half * 8;
            *(float4*)(hp)     = make_float4(acc[t][0], acc[t][1], acc[t][2], acc[t][3]);
            *(float4*)(hp + 4) = make_float4(acc[t][4], acc[t][5], acc[t][6], acc[t][7]);
        }
    }
}

// ---------------------------------------------------------------------------
// Kernel 4: sparse aggregation layer 1. 16 lanes per edge (lane j = channel).
//   h1[s][j] += val * h[p][o][j]
// ---------------------------------------------------------------------------
__global__ void agg1_kernel(const float* __restrict__ h,
                            const float* __restrict__ eval,
                            const int* __restrict__ es,
                            const int* __restrict__ ep,
                            const int* __restrict__ eo,
                            float* __restrict__ h1) {
    long long tid = (long long)blockIdx.x * blockDim.x + threadIdx.x;
    int e = (int)(tid >> 4);
    int j = (int)(tid & 15);
    if (e >= EE) return;
    int   s = es[e], p = ep[e], o = eo[e];
    float v = eval[e];
    float hv = h[((size_t)p * NN + o) * WSZ + j];
    atomicAdd(&h1[(size_t)s * WSZ + j], v * hv);
}

// ---------------------------------------------------------------------------
// Kernel 5: h1 = relu(h1 + bias1)
// ---------------------------------------------------------------------------
__global__ void bias_relu_kernel(float* __restrict__ h1, const float* __restrict__ bias1) {
    int idx = blockIdx.x * blockDim.x + threadIdx.x;
    if (idx >= NN * WSZ) return;
    float v = h1[idx] + bias1[idx & 15];
    h1[idx] = v > 0.0f ? v : 0.0f;
}

// ---------------------------------------------------------------------------
// Kernel 6: fused layer-2 projection + aggregation. 16 lanes/edge, lane c<10:
//   out[s][c] += val * sum_j h1[o][j] * W2[p][j][c]   (W2 staged in LDS)
// ---------------------------------------------------------------------------
__global__ __launch_bounds__(256) void layer2_kernel(
    const float* __restrict__ h1, const float* __restrict__ W2,
    const float* __restrict__ eval,
    const int* __restrict__ es, const int* __restrict__ ep,
    const int* __restrict__ eo, float* __restrict__ out) {
    __shared__ float sW2[RR * WSZ * CC];  // 21120 B
    for (int i = threadIdx.x; i < RR * WSZ * CC; i += blockDim.x) sW2[i] = W2[i];
    __syncthreads();

    long long tid = (long long)blockIdx.x * blockDim.x + threadIdx.x;
    int e = (int)(tid >> 4);
    int c = (int)(tid & 15);
    if (e >= EE || c >= CC) return;

    int   s = es[e], p = ep[e], o = eo[e];
    float v = eval[e];
    const float* hr = h1 + (size_t)o * WSZ;
    const float* wp = sW2 + p * (WSZ * CC);
    float a = 0.0f;
#pragma unroll
    for (int j = 0; j < WSZ; ++j) a += hr[j] * wp[j * CC + c];
    atomicAdd(&out[(size_t)s * CC + c], v * a);
}

// ---------------------------------------------------------------------------
extern "C" void kernel_launch(void* const* d_in, const int* in_sizes, int n_in,
                              void* d_out, int out_size, void* d_ws, size_t ws_size,
                              hipStream_t stream) {
    const float* X     = (const float*)d_in[0];  // [N, D]
    const float* W1    = (const float*)d_in[1];  // [R, D, WS]
    const float* bias1 = (const float*)d_in[2];  // [WS]
    const float* W2    = (const float*)d_in[3];  // [R, WS, C]
    const float* bias2 = (const float*)d_in[4];  // [C]
    const float* eval  = (const float*)d_in[5];  // [E]
    const int*   es    = (const int*)d_in[6];
    const int*   ep    = (const int*)d_in[7];
    const int*   eo    = (const int*)d_in[8];
    float* out = (float*)d_out;                  // [N, C]

    // workspace layout (offsets 256-aligned)
    char*   ws  = (char*)d_ws;
    float*  h   = (float*)ws;                           // R*N*16*4  = 105,600,000 B
    float*  h1  = (float*)(ws + 105600000);             // N*16*4    =   3,200,000 B
    __bf16* W1t = (__bf16*)(ws + 105600000 + 3200000);  // 40*16*D*2 =   2,048,000 B

    cvt_w1_kernel<<<(RPAD * DD * WSZ + 255) / 256, 256, 0, stream>>>(W1, W1t);
    zero_h1_kernel<<<(NN * WSZ + 255) / 256, 256, 0, stream>>>(h1);
    init_out_kernel<<<(NN * CC + 255) / 256, 256, 0, stream>>>(out, bias2);
    gemm1_bf16_kernel<<<NN / 16, 256, 0, stream>>>(X, W1t, h);
    {
        long long threads = (long long)EE * 16;
        agg1_kernel<<<(unsigned)((threads + 255) / 256), 256, 0, stream>>>(
            h, eval, es, ep, eo, h1);
    }
    bias_relu_kernel<<<(NN * WSZ + 255) / 256, 256, 0, stream>>>(h1, bias1);
    {
        long long threads = (long long)EE * 16;
        layer2_kernel<<<(unsigned)((threads + 255) / 256), 256, 0, stream>>>(
            h1, W2, eval, es, ep, eo, out);
    }
}
